// NISER_79104707658086
// MI455X (gfx1250) — matrix operand order repaired and checked by reading, not compiled, not run on hardware
//
#include <hip/hip_runtime.h>
#include <hip/hip_bf16.h>
#include <math.h>

// ---------------------------------------------------------------------------
// NISER forward for MI455X (gfx1250, wave32, WMMA).
// All dense matmuls use v_wmma_f32_16x16x32_f16 (f16 in, f32 accumulate).
// Weights are pre-converted to f16 once per launch so the GEMM inner loop
// loads the B operand directly as v16h (2x global_load_b128), no cvt chain.
// Big GEMM (logits = sr @ target^T, 2048x100000) fuses partial sum-exp into
// the epilogue; |logit| <= 12 so no max-subtraction is needed for a stable
// log-softmax. Peak workspace ~119 MB.
// ---------------------------------------------------------------------------

typedef __attribute__((ext_vector_type(16))) _Float16 v16h;
typedef __attribute__((ext_vector_type(8)))  float    v8f;

#define DIM 128

__device__ __forceinline__ float sig_(float x) { return 1.0f / (1.0f + __expf(-x)); }

// ---------------- elementwise / setup kernels ------------------------------

__global__ void k_zero(float* p, long long n) {
    long long i = (long long)blockIdx.x * blockDim.x + threadIdx.x;
    if (i < n) p[i] = 0.0f;
}

__global__ void k_f16cvt(const float* __restrict__ src, _Float16* __restrict__ dst, int n) {
    int i = blockIdx.x * blockDim.x + threadIdx.x;
    if (i < n) dst[i] = (_Float16)src[i];
}

// feat[j,:] = emb[iid[j],:] / (||.|| + 1e-12)   ; one block (128 thr) per row
__global__ void k_feat(const float* __restrict__ emb, const int* __restrict__ iid,
                       float* __restrict__ feat) {
    __shared__ float red[DIM];
    int j = blockIdx.x, d = threadIdx.x;
    float v = emb[(size_t)iid[j] * DIM + d];
    red[d] = v * v; __syncthreads();
    for (int s = DIM >> 1; s > 0; s >>= 1) { if (d < s) red[d] += red[d + s]; __syncthreads(); }
    feat[(size_t)j * DIM + d] = v / (sqrtf(red[0]) + 1e-12f);
}

// tgt[v,:] = f16( emb[v,:] / (||.|| + 1e-12) )  ; one block per item row
__global__ void k_target(const float* __restrict__ emb, _Float16* __restrict__ tgt) {
    __shared__ float red[DIM];
    int j = blockIdx.x, d = threadIdx.x;
    float v = emb[(size_t)j * DIM + d];
    red[d] = v * v; __syncthreads();
    for (int s = DIM >> 1; s > 0; s >>= 1) { if (d < s) red[d] += red[d + s]; __syncthreads(); }
    tgt[(size_t)j * DIM + d] = (_Float16)(v / (sqrtf(red[0]) + 1e-12f));
}

// scatter: msum[dst] += feat[src]*w ; wsum[dst] += w
__global__ void k_edge(const float* __restrict__ feat, const int* __restrict__ s,
                       const int* __restrict__ d, const float* __restrict__ w,
                       float* __restrict__ msum, float* __restrict__ wsum, int E) {
    int i = blockIdx.x * blockDim.x + threadIdx.x;
    if (i >= E * DIM) return;
    int e = i / DIM, dd = i - e * DIM;
    float we = w[e];
    atomicAdd(&msum[(size_t)d[e] * DIM + dd], feat[(size_t)s[e] * DIM + dd] * we);
    if (dd == 0) atomicAdd(&wsum[d[e]], we);
}

__global__ void k_div(float* __restrict__ msum, const float* __restrict__ wsum, int N) {
    int i = blockIdx.x * blockDim.x + threadIdx.x;
    if (i >= N * DIM) return;
    float ws = wsum[i / DIM];
    msum[i] = msum[i] / ((ws == 0.0f) ? 1.0f : ws);
}

// ---------------- generic WMMA GEMM: C[:,ccol0+] = A @ W^T (+bias) ---------
// A [M,lda] fp32, W [Nn,K] f16 row-major, C fp32 with leading dim ldc.
// One wave computes one 16x16 tile. M,Nn multiples of 16; K multiple of 32.
__global__ void k_gemm_wmma(const float* __restrict__ A, int lda,
                            const _Float16* __restrict__ W, const float* __restrict__ bias,
                            float* __restrict__ C, int ldc, int ccol0,
                            int M, int Nn, int K) {
    int wid = blockIdx.x * (blockDim.x >> 5) + (threadIdx.x >> 5);
    int ntn = Nn >> 4;
    if (wid >= (M >> 4) * ntn) return;
    int mt = wid / ntn, nt = wid - mt * ntn;
    int lane = threadIdx.x & 31, half = lane >> 4, lr = lane & 15;

    const float*    arow = A + (size_t)(mt * 16 + lr) * lda;
    const _Float16* wrow = W + (size_t)(nt * 16 + lr) * K;
    v8f acc = {};
    for (int k0 = 0; k0 < K; k0 += 32) {
        v16h a;
        const float* pa = arow + k0 + half * 16;     // A layout: halves hold K {0..15}/{16..31}
#pragma unroll
        for (int i = 0; i < 8; ++i) { a[i] = (_Float16)pa[i]; a[8 + i] = (_Float16)pa[8 + i]; }
        // B layout: lane=col, half holds contiguous 16 K -> direct 32B vector load
        v16h b = *reinterpret_cast<const v16h*>(wrow + k0 + half * 16);
        acc = __builtin_amdgcn_wmma_f32_16x16x32_f16(false, a, false, b, (short)0, acc, false, false);
    }
    float bb = bias ? bias[nt * 16 + lr] : 0.0f;
    int col = ccol0 + nt * 16 + lr;
#pragma unroll
    for (int r = 0; r < 8; ++r)
        C[(size_t)(mt * 16 + r + 8 * half) * ldc + col] = acc[r] + bb;
}

// ---------------- GRU / norms / attention ----------------------------------

__global__ void k_gru(const float* __restrict__ gi, const float* __restrict__ gh,
                      const float* __restrict__ feat, float* __restrict__ feat2, int N) {
    int i = blockIdx.x * blockDim.x + threadIdx.x;
    if (i >= N * DIM) return;
    int j = i / DIM, d = i - j * DIM;
    const float* gij = gi + (size_t)j * 3 * DIM;
    const float* ghj = gh + (size_t)j * 3 * DIM;
    float r = sig_(gij[d] + ghj[d]);
    float z = sig_(gij[DIM + d] + ghj[DIM + d]);
    float n = tanhf(gij[2 * DIM + d] + r * ghj[2 * DIM + d]);
    float h = feat[i];
    feat2[(size_t)j * 2 * DIM + d]       = (1.0f - z) * n + z * h;
    feat2[(size_t)j * 2 * DIM + DIM + d] = h;
}

// in-place row L2 norm; blockDim.x == cols (power of two)
__global__ void k_rownorm(float* __restrict__ x, int cols, float eps) {
    extern __shared__ float red[];
    int j = blockIdx.x, d = threadIdx.x;
    float v = x[(size_t)j * cols + d];
    red[d] = v * v; __syncthreads();
    for (int s = cols >> 1; s > 0; s >>= 1) { if (d < s) red[d] += red[d + s]; __syncthreads(); }
    x[(size_t)j * cols + d] = v / (sqrtf(red[0]) + eps);
}

// sr_l[b,:] = feat2[last[b],:]; also copy into srcat[:, 0:256] (ldc=384)
__global__ void k_last(const float* __restrict__ feat2, const int* __restrict__ last,
                       float* __restrict__ sr_l, float* __restrict__ srcat) {
    int b = blockIdx.x, t = threadIdx.x; // 256 threads
    float v = feat2[(size_t)last[b] * 2 * DIM + t];
    sr_l[(size_t)b * 2 * DIM + t] = v;
    srcat[(size_t)b * 3 * DIM + t] = v;
}

// e[j] = sum_d sigmoid(fu[j,d]+fv[seg[j],d]) * We[d]
__global__ void k_escore(const float* __restrict__ fu, const float* __restrict__ fv,
                         const int* __restrict__ seg, const float* __restrict__ We,
                         float* __restrict__ e, int N) {
    int j = blockIdx.x * blockDim.x + threadIdx.x;
    if (j >= N) return;
    const float* fuj = fu + (size_t)j * DIM;
    const float* fvs = fv + (size_t)seg[j] * DIM;
    float acc = 0.0f;
#pragma unroll 8
    for (int d = 0; d < DIM; ++d) acc += sig_(fuj[d] + fvs[d]) * We[d];
    e[j] = acc;
}

// segment softmax (S nodes per session) + weighted pool of feat2 -> pooled[B,256]
__global__ void k_pool(const float* __restrict__ feat2, const float* __restrict__ e,
                       float* __restrict__ pooled, int S) {
    int b = blockIdx.x, t = threadIdx.x; // 256 threads
    float m = -1e30f;
    for (int s = 0; s < S; ++s) m = fmaxf(m, e[b * S + s]);
    float sum = 0.0f;
    for (int s = 0; s < S; ++s) sum += __expf(e[b * S + s] - m);
    float acc = 0.0f;
    for (int s = 0; s < S; ++s)
        acc += (__expf(e[b * S + s] - m) / sum) * feat2[(size_t)(b * S + s) * 2 * DIM + t];
    pooled[(size_t)b * 2 * DIM + t] = acc;
}

// ---------------- big GEMM: out = 12 * sr @ tgt^T, fused partial sum-exp ---
// grid: (colGroups, B/16); block 256 = 8 waves; each wave strides n-tiles.
__global__ void k_logits(const float* __restrict__ sr, const _Float16* __restrict__ tgt,
                         float* __restrict__ out, float* __restrict__ rowsum,
                         int V, int ntPerGroup) {
    int mt = blockIdx.y;
    int wv = threadIdx.x >> 5;
    int lane = threadIdx.x & 31, half = lane >> 4, lr = lane & 15;
    int ntMax = V >> 4;
    int ntBeg = blockIdx.x * ntPerGroup;
    int ntEnd = ntBeg + ntPerGroup; if (ntEnd > ntMax) ntEnd = ntMax;

    // preload all 4 A-fragments (K=128) for this 16-row strip of sr
    v16h afr[4];
    const float* arow = sr + (size_t)(mt * 16 + lr) * DIM;
#pragma unroll
    for (int kk = 0; kk < 4; ++kk) {
        const float* pa = arow + kk * 32 + half * 16;
#pragma unroll
        for (int i = 0; i < 8; ++i) { afr[kk][i] = (_Float16)pa[i]; afr[kk][8 + i] = (_Float16)pa[8 + i]; }
    }
    float esum[8];
#pragma unroll
    for (int r = 0; r < 8; ++r) esum[r] = 0.0f;

    for (int nt = ntBeg + wv; nt < ntEnd; nt += 8) {
        const _Float16* trow = tgt + (size_t)(nt * 16 + lr) * DIM;
        __builtin_prefetch(trow + 8 * 16 * DIM, 0, 1);   // next tile for this wave
        v8f acc = {};
#pragma unroll
        for (int kk = 0; kk < 4; ++kk) {
            v16h b = *reinterpret_cast<const v16h*>(trow + kk * 32 + half * 16);
            acc = __builtin_amdgcn_wmma_f32_16x16x32_f16(false, afr[kk], false, b, (short)0, acc, false, false);
        }
        int col = nt * 16 + lr;
#pragma unroll
        for (int r = 0; r < 8; ++r) {
            float v = 12.0f * acc[r];
            out[(size_t)(mt * 16 + r + 8 * half) * V + col] = v;
            esum[r] += __expf(v);             // safe: v in [-12,12]
        }
    }
    // reduce across the 16 lanes of each half, then 2 atomics per VGPR row
#pragma unroll
    for (int r = 0; r < 8; ++r) {
        float v = esum[r];
        for (int m = 8; m >= 1; m >>= 1) v += __shfl_xor(v, m, 32);
        if (lr == 0) atomicAdd(&rowsum[mt * 16 + r + 8 * half], v);
    }
}

__global__ void k_lse(const float* __restrict__ rowsum, float* __restrict__ lse, int B) {
    int b = blockIdx.x * blockDim.x + threadIdx.x;
    if (b < B) lse[b] = logf(rowsum[b]);
}

// out[b, :] -= lse[b]   (vectorized float4; V % 4 == 0)
__global__ void k_sub(float* __restrict__ out, const float* __restrict__ lse, int V) {
    int b = blockIdx.y;
    float l = lse[b];
    int i4 = blockIdx.x * blockDim.x + threadIdx.x;
    int n4 = V >> 2;
    if (i4 < n4) {
        float4* p = (float4*)(out + (size_t)b * V);
        float4 v = p[i4];
        v.x -= l; v.y -= l; v.z -= l; v.w -= l;
        p[i4] = v;
    }
}

// ---------------------------------------------------------------------------

extern "C" void kernel_launch(void* const* d_in, const int* in_sizes, int n_in,
                              void* d_out, int out_size, void* d_ws, size_t ws_size,
                              hipStream_t stream) {
    (void)n_in; (void)out_size; (void)ws_size;
    const int*   iid   = (const int*)d_in[0];
    const int*   src   = (const int*)d_in[1];
    const int*   dst   = (const int*)d_in[2];
    const int*   seg   = (const int*)d_in[3];
    const int*   last  = (const int*)d_in[4];
    const float* w     = (const float*)d_in[5];
    const float* emb   = (const float*)d_in[6];
    const float* W1    = (const float*)d_in[7];
    const float* W2    = (const float*)d_in[8];
    const float* W_ih  = (const float*)d_in[9];
    const float* W_hh  = (const float*)d_in[10];
    const float* b_ih  = (const float*)d_in[11];
    const float* b_hh  = (const float*)d_in[12];
    const float* Wu    = (const float*)d_in[13];
    const float* Wv    = (const float*)d_in[14];
    const float* bv    = (const float*)d_in[15];
    const float* We    = (const float*)d_in[16];
    const float* Wout  = (const float*)d_in[17];
    const float* Wsr   = (const float*)d_in[18];
    float* out = (float*)d_out;

    const int N = in_sizes[0];                 // 16384 nodes
    const int E = in_sizes[1];                 // 14336 edges
    const int B = in_sizes[4];                 // 2048 sessions
    const int V = in_sizes[6] / DIM;           // 100000 items
    const int S = N / B;                       // 8 nodes/session

    // ---- workspace layout (lifetime-aliased); all region sizes 256B-multiples
    size_t off = 0;
    auto alloc = [&](size_t bytes) { size_t o = off; off += (bytes + 255) & ~(size_t)255; return o; };
    size_t off_tgt   = alloc((size_t)V * DIM * sizeof(_Float16)); // 25.6 MB, persists
    size_t off_feat  = alloc((size_t)N * DIM * 4);                // 8.4 MB
    size_t off_msum1 = alloc((size_t)N * DIM * 4);                // 8.4 MB
    size_t off_msum2 = alloc((size_t)N * DIM * 4);                // 8.4 MB (contig after msum1)
    size_t off_wsum1 = alloc((size_t)N * 4);
    size_t off_wsum2 = alloc((size_t)N * 4);
    size_t off_xcat  = alloc((size_t)N * 2 * DIM * 4);            // 16.8 MB
    size_t off_gi    = alloc((size_t)N * 3 * DIM * 4);            // 25.2 MB
    size_t off_gh    = alloc((size_t)N * 3 * DIM * 4);            // 25.2 MB
    // f16 copies of weights (persist whole launch, ~650 KB total)
    size_t off_hW1   = alloc((size_t)DIM * DIM * 2);
    size_t off_hW2   = alloc((size_t)DIM * DIM * 2);
    size_t off_hWih  = alloc((size_t)3 * DIM * 2 * DIM * 2);
    size_t off_hWhh  = alloc((size_t)3 * DIM * DIM * 2);
    size_t off_hWu   = alloc((size_t)DIM * 2 * DIM * 2);
    size_t off_hWv   = alloc((size_t)DIM * 2 * DIM * 2);
    size_t off_hWout = alloc((size_t)DIM * 2 * DIM * 2);
    size_t off_hWsr  = alloc((size_t)DIM * 3 * DIM * 2);
    // aliases (safe by lifetime):
    size_t off_feat2  = off_msum1;                                // [N,256] over msum1+msum2
    size_t off_fu     = off_xcat;                                 // [N,128]
    size_t off_fv     = off_fu + (size_t)N * DIM * 4;             // [B,128]
    size_t off_e      = off_fv + (size_t)B * DIM * 4;             // [N]
    size_t off_srl    = off_e  + (size_t)N * 4;                   // [B,256]
    size_t off_srcat  = off_gi;                                   // [B,384]
    size_t off_pooled = off_srcat  + (size_t)B * 3 * DIM * 4;     // [B,256]
    size_t off_sr     = off_pooled + (size_t)B * 2 * DIM * 4;     // [B,128]
    size_t off_rowsum = off_sr     + (size_t)B * DIM * 4;         // [B]
    size_t off_lse    = off_rowsum + (size_t)B * 4;               // [B]

    char* ws = (char*)d_ws;
    auto F  = [&](size_t o) { return (float*)(ws + o); };
    auto H  = [&](size_t o) { return (_Float16*)(ws + o); };

    auto gemm = [&](const float* A, int lda, const _Float16* Wm, const float* bias,
                    float* C, int ldc, int ccol0, int M, int Nn, int K) {
        int tiles = (M >> 4) * (Nn >> 4);
        k_gemm_wmma<<<(tiles + 7) / 8, 256, 0, stream>>>(A, lda, Wm, bias, C, ldc, ccol0, M, Nn, K);
    };
    auto cvt = [&](const float* src_, size_t dsto, int n) {
        k_f16cvt<<<(n + 255) / 256, 256, 0, stream>>>(src_, H(dsto), n);
    };

    // 0) zero accumulators (msum1,msum2,wsum1,wsum2 are contiguous: N*(2*128+2) floats)
    long long nz = (long long)N * (2 * DIM + 2);
    k_zero<<<(unsigned)((nz + 255) / 256), 256, 0, stream>>>(F(off_msum1), nz);
    k_zero<<<(B + 255) / 256, 256, 0, stream>>>(F(off_rowsum), B);

    // 0b) weights -> f16 (once per launch)
    cvt(W1,   off_hW1,   DIM * DIM);
    cvt(W2,   off_hW2,   DIM * DIM);
    cvt(W_ih, off_hWih,  3 * DIM * 2 * DIM);
    cvt(W_hh, off_hWhh,  3 * DIM * DIM);
    cvt(Wu,   off_hWu,   DIM * 2 * DIM);
    cvt(Wv,   off_hWv,   DIM * 2 * DIM);
    cvt(Wout, off_hWout, DIM * 2 * DIM);
    cvt(Wsr,  off_hWsr,  DIM * 3 * DIM);

    // 1) feat = norm(emb[iid]); target (f16) = norm(emb)
    k_feat<<<N, DIM, 0, stream>>>(emb, iid, F(off_feat));
    k_target<<<V, DIM, 0, stream>>>(emb, H(off_tgt));

    // 2) weighted segment means, both directions
    int et = E * DIM;
    k_edge<<<(et + 255) / 256, 256, 0, stream>>>(F(off_feat), src, dst, w, F(off_msum1), F(off_wsum1), E);
    k_edge<<<(et + 255) / 256, 256, 0, stream>>>(F(off_feat), dst, src, w, F(off_msum2), F(off_wsum2), E);
    int ndt = N * DIM;
    k_div<<<(ndt + 255) / 256, 256, 0, stream>>>(F(off_msum1), F(off_wsum1), N);
    k_div<<<(ndt + 255) / 256, 256, 0, stream>>>(F(off_msum2), F(off_wsum2), N);

    // 3) xcat = [n1raw@W1^T , n2raw@W2^T]
    gemm(F(off_msum1), DIM, H(off_hW1), nullptr, F(off_xcat), 2 * DIM, 0,   N, DIM, DIM);
    gemm(F(off_msum2), DIM, H(off_hW2), nullptr, F(off_xcat), 2 * DIM, DIM, N, DIM, DIM);

    // 4) GRU gates
    gemm(F(off_xcat), 2 * DIM, H(off_hWih), b_ih, F(off_gi), 3 * DIM, 0, N, 3 * DIM, 2 * DIM);
    gemm(F(off_feat),     DIM, H(off_hWhh), b_hh, F(off_gh), 3 * DIM, 0, N, 3 * DIM, DIM);
    k_gru<<<(ndt + 255) / 256, 256, 0, stream>>>(F(off_gi), F(off_gh), F(off_feat), F(off_feat2), N);
    k_rownorm<<<N, 2 * DIM, 2 * DIM * 4, stream>>>(F(off_feat2), 2 * DIM, 0.0f);

    // 5) attention readout
    gemm(F(off_feat2), 2 * DIM, H(off_hWu), nullptr, F(off_fu), DIM, 0, N, DIM, 2 * DIM);
    k_last<<<B, 2 * DIM, 0, stream>>>(F(off_feat2), last, F(off_srl), F(off_srcat));
    gemm(F(off_srl), 2 * DIM, H(off_hWv), bv, F(off_fv), DIM, 0, B, DIM, 2 * DIM);
    k_escore<<<(N + 255) / 256, 256, 0, stream>>>(F(off_fu), F(off_fv), seg, We, F(off_e), N);
    k_pool<<<B, 2 * DIM, 0, stream>>>(F(off_feat2), F(off_e), F(off_pooled), S);
    gemm(F(off_pooled), 2 * DIM, H(off_hWout), nullptr, F(off_srcat), 3 * DIM, 2 * DIM, B, DIM, 2 * DIM);

    // 6) session representation
    gemm(F(off_srcat), 3 * DIM, H(off_hWsr), nullptr, F(off_sr), DIM, 0, B, DIM, 3 * DIM);
    k_rownorm<<<B, DIM, DIM * 4, stream>>>(F(off_sr), DIM, 1e-12f);

    // 7) logits + fused sum-exp; then log-softmax correction
    int ntiles = V >> 4;                       // 6250
    int groups = 32;
    int ntPer = (ntiles + groups - 1) / groups;
    dim3 lg(groups, B >> 4);
    k_logits<<<lg, 256, 0, stream>>>(F(off_sr), H(off_tgt), out, F(off_rowsum), V, ntPer);
    k_lse<<<(B + 255) / 256, 256, 0, stream>>>(F(off_rowsum), F(off_lse), B);
    dim3 sg(((V >> 2) + 255) / 256, B);
    k_sub<<<sg, 256, 0, stream>>>(out, F(off_lse), V);
}
